// MKMGCN_8753143349540
// MI455X (gfx1250) — matrix operand order repaired and checked by reading, not compiled, not run on hardware
//
#include <hip/hip_runtime.h>
#include <hip/hip_bf16.h>
#include <math.h>

#define N_NODES 100000
#define N_EDGES 1600000
#define IN_DIM  1024
#define HID     64
#define N_PAIRS 4096

typedef __attribute__((ext_vector_type(16))) __bf16 v16bf;
typedef __attribute__((ext_vector_type(8)))  float  v8f;

union ABFrag { unsigned int u[8]; v16bf v; };

__device__ __forceinline__ unsigned short f2bf(float f) {
    unsigned int u = __float_as_uint(f);
    u += 0x7FFFu + ((u >> 16) & 1u);   // round-to-nearest-even
    return (unsigned short)(u >> 16);
}

#if __has_builtin(__builtin_amdgcn_cvt_pk_bf16_f32)
__device__ __forceinline__ unsigned int pk2bf(float a, float b) {
    auto r = __builtin_amdgcn_cvt_pk_bf16_f32(a, b);   // v_cvt_pk_bf16_f32
    unsigned int u; __builtin_memcpy(&u, &r, 4); return u;
}
#else
__device__ __forceinline__ unsigned int pk2bf(float a, float b) {
    return (unsigned int)f2bf(a) | ((unsigned int)f2bf(b) << 16);
}
#endif

#if __has_builtin(__builtin_amdgcn_global_load_async_to_lds_b128)
#define ASYNC_LDS 1
typedef int v4i_ __attribute__((vector_size(16)));
typedef __attribute__((address_space(1))) v4i_* as1_v4p;   // global (AS1)
typedef __attribute__((address_space(3))) v4i_* as3_v4p;   // LDS (AS3)
__device__ __forceinline__ void async_copy16(const void* g, void* l) {
    __builtin_amdgcn_global_load_async_to_lds_b128(
        (as1_v4p)(unsigned long long)(uintptr_t)g,
        (as3_v4p)(unsigned)(uintptr_t)l,
        0, 0);
}
__device__ __forceinline__ void wait_async0() {
#if __has_builtin(__builtin_amdgcn_s_wait_asynccnt)
    __builtin_amdgcn_s_wait_asynccnt(0);
#else
    asm volatile("s_wait_asynccnt 0x0" ::: "memory");
#endif
}
#else
#define ASYNC_LDS 0
#endif

__device__ __forceinline__ float frelu(float v) { return v > 0.f ? v : 0.f; }

// ---------------- zero scratch (float4) ----------------
__global__ void k_zero(float* __restrict__ p, long long n4) {
    long long i = (long long)blockIdx.x * blockDim.x + threadIdx.x;
    long long st = (long long)gridDim.x * blockDim.x;
    float4 z = make_float4(0.f, 0.f, 0.f, 0.f);
    for (; i < n4; i += st) ((float4*)p)[i] = z;
}

// ---------------- one-time bf16 transposed weights ----------------
__global__ void k_prep_w(const float* __restrict__ Wl, const float* __restrict__ Wh,
                         const float* __restrict__ embW, const float* __restrict__ W1,
                         unsigned short* __restrict__ WtX, unsigned short* __restrict__ WtE,
                         unsigned short* __restrict__ WtD) {
    int i = blockIdx.x * blockDim.x + threadIdx.x;
    int st = gridDim.x * blockDim.x;
    for (int j = i; j < IN_DIM * 128; j += st) {          // WtX[n][k]
        int k = j >> 7, n = j & 127;
        float v = (n < HID) ? Wl[k * HID + n] : Wh[k * HID + (n - HID)];
        WtX[n * IN_DIM + k] = f2bf(v);
    }
    for (int j = i; j < 128 * HID; j += st) {             // WtE/WtD[n][k]
        int k = j >> 6, n = j & 63;
        WtE[n * 128 + k] = f2bf(embW[k * HID + n]);
        WtD[n * 128 + k] = f2bf(W1[k * HID + n]);
    }
}

// ---------------- degree histograms ----------------
__global__ void k_deg(const int* __restrict__ ei,
                      float* __restrict__ deg_out, float* __restrict__ deg_in) {
    int i = blockIdx.x * blockDim.x + threadIdx.x;
    int st = gridDim.x * blockDim.x;
    for (; i < N_EDGES; i += st) {
        atomicAdd(&deg_out[ei[i]], 1.0f);
        atomicAdd(&deg_in[ei[N_EDGES + i]], 1.0f);
    }
}

// ------- h = x @ [W_low | W_high]  (bf16 WMMA, 32x128 tile, double-buffered) -
__global__ __launch_bounds__(256) void k_gemm_x(const float* __restrict__ x,
        const unsigned short* __restrict__ Wt,   // [128][1024] bf16
        float* __restrict__ h) {
    __shared__ __align__(16) unsigned short sA[2][32][40];   // 80B pitch
    __shared__ __align__(16) unsigned short sB[2][128][40];
    const int t = threadIdx.x;
    const int w = t >> 5, lane = t & 31;
    const int half = lane >> 4, mn = lane & 15;
    const int m0 = blockIdx.x * 32;
    const unsigned int* uA = (const unsigned int*)&sA[0][0][0];
    const unsigned int* uB = (const unsigned int*)&sB[0][0][0];
    v8f acc0 = {}, acc1 = {};

    const int arow = t >> 3, akq = (t & 7) * 4;              // 256 float4 chunks
    const float* pxa = x + (long long)(m0 + arow) * IN_DIM + akq;
    const int bn = t >> 2, bo8 = (t & 3) * 8;                // B chunk coords

    auto fillB = [&](int buf, int k0) {
#pragma unroll
        for (int j = 0; j < 2; ++j) {
            int n = bn + 64 * j;
#if ASYNC_LDS
            async_copy16(Wt + n * IN_DIM + k0 + bo8, &sB[buf][n][bo8]);
#else
            *(uint4*)&sB[buf][n][bo8] = *(const uint4*)(Wt + n * IN_DIM + k0 + bo8);
#endif
        }
    };
    auto storeA = [&](int buf, const float4& v) {
        uint2 p; p.x = pk2bf(v.x, v.y); p.y = pk2bf(v.z, v.w);
        *(uint2*)&sA[buf][arow][akq] = p;
    };

    {   // prologue: stage tile 0
        const float4 v0 = *(const float4*)(pxa);
        fillB(0, 0);
        storeA(0, v0);
    }
    int cb = 0;
    for (int k0 = 0; k0 < IN_DIM; k0 += 32) {
#if ASYNC_LDS
        wait_async0();
#endif
        __syncthreads();
        // (1) fragment LDS reads issue first
        ABFrag a0, a1, b;
        const unsigned abase = (unsigned)cb * 640u;          // 32*40/2 uints
        const unsigned ia0 = abase + (unsigned)mn * 20u + 4u * half;
        const unsigned ia1 = ia0 + 16u * 20u;
#pragma unroll
        for (int j = 0; j < 4; ++j) {
            a0.u[j] = uA[ia0 + j]; a0.u[4 + j] = uA[ia0 + 8 + j];
            a1.u[j] = uA[ia1 + j]; a1.u[4 + j] = uA[ia1 + 8 + j];
        }
        const unsigned ib = (unsigned)cb * 2560u             // 128*40/2 uints
                          + (unsigned)(w * 16 + mn) * 20u + 4u * half;
#pragma unroll
        for (int j = 0; j < 4; ++j) { b.u[j] = uB[ib + j]; b.u[4 + j] = uB[ib + 8 + j]; }
        // (2) next-tile memory ops issue before compute
        const bool more = (k0 + 32 < IN_DIM);
        float4 va = make_float4(0.f, 0.f, 0.f, 0.f);
        if (more) {
            va = *(const float4*)(pxa + k0 + 32);
            __builtin_prefetch(pxa + k0 + 64, 0, 0);
            fillB(cb ^ 1, k0 + 32);
        }
        // (3) matrix ops (XDL pipe; VALU conversion below co-executes)
        acc0 = __builtin_amdgcn_wmma_f32_16x16x32_bf16(false, a0.v, false, b.v,
                                                       (short)0, acc0, false, false);
        acc1 = __builtin_amdgcn_wmma_f32_16x16x32_bf16(false, a1.v, false, b.v,
                                                       (short)0, acc1, false, false);
        // (4) convert + stage A for next tile after WMMA issue
        if (more) storeA(cb ^ 1, va);
        cb ^= 1;
    }
    const int col = w * 16 + mn;
#pragma unroll
    for (int r = 0; r < 8; ++r) {
        h[(long long)(m0 + r + 8 * half) * 128 + col]      = acc0[r];
        h[(long long)(m0 + 16 + r + 8 * half) * 128 + col] = acc1[r];
    }
}

// ---------------- agg[dst] += norm * h[src]  (scatter-add) -------------------
__global__ __launch_bounds__(256) void k_scatter(const int* __restrict__ ei,
        const float* __restrict__ h, const float* __restrict__ deg_out,
        const float* __restrict__ deg_in, float* __restrict__ agg) {
    long long gid = (long long)blockIdx.x * blockDim.x + threadIdx.x;
    long long nth = (long long)gridDim.x * blockDim.x;
    const long long total = (long long)N_EDGES * 32;
    for (long long i = gid; i < total; i += nth) {
        int e = (int)(i >> 5), lane = (int)(i & 31);
        int src = ei[e], dst = ei[N_EDGES + e];
        float no = deg_out[src]; no = no < 1.f ? 1.f : no;
        float ni = deg_in[dst];  ni = ni < 1.f ? 1.f : ni;
        float norm = rsqrtf(no) * rsqrtf(ni);
        const float4 v = *(const float4*)&h[(long long)src * 128 + lane * 4];
        float* base = &agg[(long long)dst * 128 + lane * 4];
        atomicAdd(base + 0, norm * v.x);
        atomicAdd(base + 1, norm * v.y);
        atomicAdd(base + 2, norm * v.z);
        atomicAdd(base + 3, norm * v.w);
    }
}

// ------- emb = relu(fused @ emb_W + b); SSL stats (32-row macro-tile) --------
__global__ __launch_bounds__(128) void k_post_emb(const float* __restrict__ h,
        const float* __restrict__ agg, const unsigned short* __restrict__ WtE,
        const float* __restrict__ embB, float* __restrict__ emb,
        float* __restrict__ colsum, float* __restrict__ colsumsq) {
    __shared__ __align__(16) unsigned short sA[32][40];
    __shared__ __align__(16) unsigned short sB[64][40];
    const int t = threadIdx.x;
    const int w = t >> 5, lane = t & 31;
    const int half = lane >> 4, mn = lane & 15;
    const int m0 = blockIdx.x * 32;
    const unsigned int* uA = (const unsigned int*)&sA[0][0];
    const unsigned int* uB = (const unsigned int*)&sB[0][0];
    v8f acc0 = {}, acc1 = {};
    for (int k0 = 0; k0 < 128; k0 += 32) {
        __syncthreads();
#pragma unroll
        for (int j = 0; j < 2; ++j) {                     // B tile copy (async)
            int c = t + 128 * j;
            int n = c >> 2, o8 = (c & 3) * 8;
#if ASYNC_LDS
            async_copy16(WtE + n * 128 + k0 + o8, &sB[n][o8]);
#else
            *(uint4*)&sB[n][o8] = *(const uint4*)(WtE + n * 128 + k0 + o8);
#endif
        }
#pragma unroll
        for (int j = 0; j < 2; ++j) {                     // A: fused features
            int c = t + 128 * j;
            int row = c >> 3, kq = (c & 7) * 4, kk = k0 + kq;
            long long off = (long long)(m0 + row) * 128 + kk;
            float4 va = *(const float4*)&agg[off];
            float4 vv;
            if (kk < HID) { vv = va; }
            else {
                float4 vh = *(const float4*)&h[off];
                vv.x = vh.x - va.x; vv.y = vh.y - va.y;
                vv.z = vh.z - va.z; vv.w = vh.w - va.w;
            }
            uint2 p;
            p.x = pk2bf(frelu(vv.x), frelu(vv.y));
            p.y = pk2bf(frelu(vv.z), frelu(vv.w));
            *(uint2*)&sA[row][kq] = p;
        }
#if ASYNC_LDS
        wait_async0();
#endif
        __syncthreads();
        ABFrag a0, a1, b;
        const unsigned ia0 = (unsigned)mn * 20u + 4u * half;
        const unsigned ia1 = ia0 + 16u * 20u;
#pragma unroll
        for (int j = 0; j < 4; ++j) {
            a0.u[j] = uA[ia0 + j]; a0.u[4 + j] = uA[ia0 + 8 + j];
            a1.u[j] = uA[ia1 + j]; a1.u[4 + j] = uA[ia1 + 8 + j];
        }
        const unsigned ib = (unsigned)(w * 16 + mn) * 20u + 4u * half;
#pragma unroll
        for (int j = 0; j < 4; ++j) { b.u[j] = uB[ib + j]; b.u[4 + j] = uB[ib + 8 + j]; }
        acc0 = __builtin_amdgcn_wmma_f32_16x16x32_bf16(false, a0.v, false, b.v,
                                                       (short)0, acc0, false, false);
        acc1 = __builtin_amdgcn_wmma_f32_16x16x32_bf16(false, a1.v, false, b.v,
                                                       (short)0, acc1, false, false);
    }
    const int col = w * 16 + mn;
    const float bias = embB[col];
    float s = 0.f, q = 0.f;
#pragma unroll
    for (int r = 0; r < 8; ++r) {
        float v0 = frelu(acc0[r] + bias);
        float v1 = frelu(acc1[r] + bias);
        emb[(long long)(m0 + r + 8 * half) * HID + col]      = v0;
        emb[(long long)(m0 + 16 + r + 8 * half) * HID + col] = v1;
        s += v0 + v1; q += v0 * v0 + v1 * v1;
    }
    s += __shfl_xor(s, 16, 32);
    q += __shfl_xor(q, 16, 32);
    if (half == 0) { atomicAdd(&colsum[col], s); atomicAdd(&colsumsq[col], q); }
}

// ---------------- detection MLP: WMMA layer1 + per-row dot layer2 ------------
__global__ __launch_bounds__(128) void k_pairs(const float* __restrict__ emb,
        const int* __restrict__ idxA, const int* __restrict__ idxB,
        const unsigned short* __restrict__ WtD, const float* __restrict__ b1,
        const float* __restrict__ W2, const float* __restrict__ b2,
        float* __restrict__ out) {
    __shared__ __align__(16) unsigned short sA[16][40];
    __shared__ __align__(16) unsigned short sB[64][40];
    __shared__ float sH[16][68];
    const int t = threadIdx.x;
    const int w = t >> 5, lane = t & 31;
    const int half = lane >> 4, mn = lane & 15;
    const int p0 = blockIdx.x * 16;
    const unsigned int* uA = (const unsigned int*)&sA[0][0];
    const unsigned int* uB = (const unsigned int*)&sB[0][0];
    v8f acc = {};
    for (int k0 = 0; k0 < 128; k0 += 32) {
        __syncthreads();
#pragma unroll
        for (int j = 0; j < 2; ++j) {                     // B tile copy (async)
            int c = t + 128 * j;
            int n = c >> 2, o8 = (c & 3) * 8;
#if ASYNC_LDS
            async_copy16(WtD + n * 128 + k0 + o8, &sB[n][o8]);
#else
            *(uint4*)&sB[n][o8] = *(const uint4*)(WtD + n * 128 + k0 + o8);
#endif
        }
        {   // A: gathered embeddings, one float4 per thread
            int row = t >> 3, kq = (t & 7) * 4, kk = k0 + kq;
            int node = (kk < HID) ? idxA[p0 + row] : idxB[p0 + row];
            const float4 v = *(const float4*)&emb[(long long)node * HID + (kk & (HID - 1))];
            uint2 p; p.x = pk2bf(v.x, v.y); p.y = pk2bf(v.z, v.w);
            *(uint2*)&sA[row][kq] = p;
        }
#if ASYNC_LDS
        wait_async0();
#endif
        __syncthreads();
        ABFrag a, b;
        const unsigned ia = (unsigned)mn * 20u + 4u * half;
#pragma unroll
        for (int j = 0; j < 4; ++j) { a.u[j] = uA[ia + j]; a.u[4 + j] = uA[ia + 8 + j]; }
        const unsigned ib = (unsigned)(w * 16 + mn) * 20u + 4u * half;
#pragma unroll
        for (int j = 0; j < 4; ++j) { b.u[j] = uB[ib + j]; b.u[4 + j] = uB[ib + 8 + j]; }
        acc = __builtin_amdgcn_wmma_f32_16x16x32_bf16(false, a.v, false, b.v,
                                                      (short)0, acc, false, false);
    }
    const int col = w * 16 + mn;
    const float bias = b1[col];
#pragma unroll
    for (int r = 0; r < 8; ++r) sH[r + 8 * half][col] = frelu(acc[r] + bias);
    __syncthreads();
    if (t < 16) {
        float a2 = b2[0];
#pragma unroll 8
        for (int c = 0; c < HID; ++c) a2 += sH[t][c] * W2[c];
        out[p0 + t] = 1.f / (1.f + expf(-a2));
    }
}

// ---------------- ssl scalar -------------------------------------------------
__global__ void k_ssl(const float* __restrict__ colsum,
                      const float* __restrict__ colsumsq, float* __restrict__ o) {
    __shared__ float red[64];
    int t = threadIdx.x;
    float s = colsum[t], q = colsumsq[t];
    red[t] = q - s * s * (1.0f / (float)N_NODES);
    __syncthreads();
    for (int off = 32; off > 0; off >>= 1) {
        if (t < off) red[t] += red[t + off];
        __syncthreads();
    }
    if (t == 0) o[0] = red[0] / ((float)N_NODES * (float)HID);
}

extern "C" void kernel_launch(void* const* d_in, const int* in_sizes, int n_in,
                              void* d_out, int out_size, void* d_ws, size_t ws_size,
                              hipStream_t stream) {
    const float* x    = (const float*)d_in[0];
    const float* Wl   = (const float*)d_in[1];
    const float* Wh   = (const float*)d_in[2];
    const float* embW = (const float*)d_in[3];
    const float* embB = (const float*)d_in[4];
    const float* W1   = (const float*)d_in[5];
    const float* b1   = (const float*)d_in[6];
    const float* W2   = (const float*)d_in[7];
    const float* b2   = (const float*)d_in[8];
    const int*   ei   = (const int*)d_in[9];
    const int*   dA   = (const int*)d_in[10];
    const int*   dB   = (const int*)d_in[11];
    float* out = (float*)d_out;

    float* ws       = (float*)d_ws;
    float* h        = ws;                                   // [N,128]
    float* emb      = h   + (long long)N_NODES * 128;       // [N,64]
    float* agg      = emb + (long long)N_NODES * HID;       // [N,128]  (zeroed)
    float* deg_out  = agg + (long long)N_NODES * 128;       // [N]      (zeroed)
    float* deg_in   = deg_out + N_NODES;                    // [N]      (zeroed)
    float* colsum   = deg_in  + N_NODES;                    // [64]     (zeroed)
    float* colsumsq = colsum  + HID;                        // [64]     (zeroed)
    unsigned short* WtX = (unsigned short*)(colsumsq + HID);// [128][1024] bf16
    unsigned short* WtE = WtX + 128 * IN_DIM;               // [64][128]  bf16
    unsigned short* WtD = WtE + HID * 128;                  // [64][128]  bf16

    const long long zero_n4 = ((long long)N_NODES * 128 + 2LL * N_NODES + 2 * HID) / 4;
    k_prep_w<<<512, 256, 0, stream>>>(Wl, Wh, embW, W1, WtX, WtE, WtD);
    k_zero<<<2048, 256, 0, stream>>>(agg, zero_n4);
    k_deg<<<(N_EDGES + 255) / 256, 256, 0, stream>>>(ei, deg_out, deg_in);
    k_gemm_x<<<N_NODES / 32, 256, 0, stream>>>(x, WtX, h);
    k_scatter<<<200000, 256, 0, stream>>>(ei, h, deg_out, deg_in, agg);
    k_post_emb<<<N_NODES / 32, 128, 0, stream>>>(h, agg, WtE, embB, emb,
                                                 colsum, colsumsq);
    k_pairs<<<N_PAIRS / 16, 128, 0, stream>>>(emb, dA, dB, WtD, b1, W2, b2, out);
    k_ssl<<<1, 64, 0, stream>>>(colsum, colsumsq, out + N_PAIRS);
}
